// MoEBase_68023692034150
// MI455X (gfx1250) — matrix operand order repaired and checked by reading, not compiled, not run on hardware
//
#include <hip/hip_runtime.h>

// ---------------- problem constants ----------------
#define TKN  4096   // T = B*S
#define DIM  2048   // D
#define NEXP 16     // E
#define HEXP 1024   // H (per-expert hidden)
#define HSH  4096   // HS (shared hidden)
#define TOPK 4

typedef __attribute__((ext_vector_type(4)))  float  v4f;
typedef __attribute__((ext_vector_type(8)))  float  v8f;
typedef __attribute__((ext_vector_type(16))) __bf16 v16bf;
typedef __attribute__((ext_vector_type(8)))  __bf16 v8bf;
typedef __attribute__((ext_vector_type(4)))  int    v4i;

// LDS tile strides (halves). 40*2B = 80B rows keep 16B alignment.
#define LDA 40
#define LDB 40
#define GLD 1032   // 1024 + 8 pad; 1032*2B is 16B-aligned

// ---- CDNA5 async global->LDS (ASYNCcnt) with sync fallback ----
#if defined(__has_builtin)
#if __has_builtin(__builtin_amdgcn_global_load_async_to_lds_b128) && \
    __has_builtin(__builtin_amdgcn_s_wait_asynccnt)
#define USE_ASYNC_LDS 1
#endif
#endif
#ifndef USE_ASYNC_LDS
#define USE_ASYNC_LDS 0
#endif

// builtin signature (from hipcc diagnostic): (v4i addrspace(1)*, v4i addrspace(3)*, Imm, Imm)
typedef __attribute__((address_space(1))) v4i gv4i;
typedef __attribute__((address_space(3))) v4i lv4i;

__device__ __forceinline__ void copy16_g2l(const void* g, void* l) {
#if USE_ASYNC_LDS
  __builtin_amdgcn_global_load_async_to_lds_b128((gv4i*)g, (lv4i*)l, 0, 0);
#else
  *(uint4*)l = *(const uint4*)g;
#endif
}
__device__ __forceinline__ void async_wait() {
#if USE_ASYNC_LDS
  __builtin_amdgcn_s_wait_asynccnt(0);
#endif
}

__device__ __forceinline__ unsigned pack_bf16x2(float a, float b) {
  unsigned short ua = __builtin_bit_cast(unsigned short, (__bf16)a);
  unsigned short ub = __builtin_bit_cast(unsigned short, (__bf16)b);
  return (unsigned)ua | ((unsigned)ub << 16);
}

__device__ __forceinline__ float silu_f(float v) {
#if defined(__has_builtin) && __has_builtin(__builtin_amdgcn_rcpf)
  return v * __builtin_amdgcn_rcpf(1.0f + __expf(-v));
#else
  return v / (1.0f + __expf(-v));
#endif
}

__device__ __forceinline__ v8f wmma_bf16(v16bf a, v16bf b, v8f c) {
  return __builtin_amdgcn_wmma_f32_16x16x32_bf16(false, a, false, b, (short)0, c,
                                                 false, false);
}

// A fragment (16x32 bf16, M x K): lane<16: row M=lane, K=0..7 & 16..23;
// lane>=16: row M=lane-16, K=8..15 & 24..31.
__device__ __forceinline__ v16bf load_a_frag(const __bf16* rowPtr, int lane) {
  const int base = (lane >> 4) << 3;
  v8bf lo = *(const v8bf*)(rowPtr + base);
  v8bf hi = *(const v8bf*)(rowPtr + base + 16);
  v16bf a;
#pragma unroll
  for (int i = 0; i < 8; ++i) { a[i] = lo[i]; a[i + 8] = hi[i]; }
  return a;
}

// B fragment (32x16 bf16, K x N) from N-major LDS tile.
__device__ __forceinline__ v16bf load_b_frag(const __bf16* colPtr, int lane) {
  const int base = (lane >> 4) << 3;
  v8bf lo = *(const v8bf*)(colPtr + base);
  v8bf hi = *(const v8bf*)(colPtr + base + 16);
  v16bf b;
#pragma unroll
  for (int i = 0; i < 8; ++i) { b[i] = lo[i]; b[i + 8] = hi[i]; }
  return b;
}

// ---------------- fp32 -> bf16 convert of x (packed 16B stores) -------------
__global__ __launch_bounds__(256) void k_cvt_bf16(const float* __restrict__ x,
                                                  __bf16* __restrict__ xbf) {
  const int i = (blockIdx.x * 256 + threadIdx.x) * 8;
  v4f a = *(const v4f*)(x + i);
  v4f b = *(const v4f*)(x + i + 4);
  uint4 o;
  o.x = pack_bf16x2(a[0], a[1]);
  o.y = pack_bf16x2(a[2], a[3]);
  o.z = pack_bf16x2(b[0], b[1]);
  o.w = pack_bf16x2(b[2], b[3]);
  *(uint4*)(xbf + i) = o;
}

// ---------------- router ----------------------------------------------------
__global__ __launch_bounds__(32) void k_router(const float* __restrict__ x,
                                               const float* __restrict__ wr,
                                               float* __restrict__ cw) {
  const int t    = blockIdx.x;
  const int lane = threadIdx.x;
  const int e    = lane & 15;
  const int half = lane >> 4;
  const float* xr = x + (size_t)t * DIM;

  float acc = 0.0f;
  const int d0 = half * (DIM / 2), d1 = d0 + (DIM / 2);
  for (int d = d0; d < d1; d += 4) {
    acc += xr[d + 0] * wr[(d + 0) * NEXP + e];
    acc += xr[d + 1] * wr[(d + 1) * NEXP + e];
    acc += xr[d + 2] * wr[(d + 2) * NEXP + e];
    acc += xr[d + 3] * wr[(d + 3) * NEXP + e];
  }
  acc += __shfl_xor(acc, 16, 32);

  float m = acc;
#pragma unroll
  for (int off = 8; off >= 1; off >>= 1) m = fmaxf(m, __shfl_xor(m, off, 32));
  float p = __expf(acc - m);
  float s = p;
#pragma unroll
  for (int off = 8; off >= 1; off >>= 1) s += __shfl_xor(s, off, 32);
  const float prob = p / s;

  float rem = prob, topsum = 0.0f;
  bool  sel = false;
#pragma unroll
  for (int it = 0; it < TOPK; ++it) {
    float v = rem; int vi = e;
#pragma unroll
    for (int off = 8; off >= 1; off >>= 1) {
      float ov = __shfl_xor(v, off, 32);
      int   oi = __shfl_xor(vi, off, 32);
      if (ov > v || (ov == v && oi < vi)) { v = ov; vi = oi; }
    }
    topsum += v;
    if (e == vi) { sel = true; rem = -1.0f; }
  }
  if (lane < 16) cw[(size_t)t * NEXP + e] = sel ? (prob / topsum) : 0.0f;
}

// ---------------- deterministic per-expert token-list compaction ------------
__global__ __launch_bounds__(256) void k_build_lists(const float* __restrict__ cw,
                                                     int* __restrict__ counts,
                                                     int* __restrict__ tlist) {
  const int e = blockIdx.x;
  __shared__ int wsum[8];
  __shared__ int wbase[8];
  __shared__ int running;
  const int tid = threadIdx.x, lane = tid & 31, wv = tid >> 5;
  if (tid == 0) running = 0;
  __syncthreads();
  for (int base = 0; base < TKN; base += 256) {
    const int t = base + tid;
    const bool sel = cw[(size_t)t * NEXP + e] != 0.0f;
    unsigned long long mask = __ballot(sel);
    const int pre = __popcll(mask & ((1ull << lane) - 1ull));
    if (lane == 0) wsum[wv] = __popcll(mask);
    __syncthreads();
    if (tid == 0) {
      int b = running;
      for (int i = 0; i < 8; ++i) { wbase[i] = b; b += wsum[i]; }
      running = b;
    }
    __syncthreads();
    if (sel) tlist[(size_t)e * TKN + wbase[wv] + pre] = t;
    __syncthreads();
  }
  if (tid == 0) counts[e] = running;
}

// ---------------- shared MLP up (double-buffered) ---------------------------
__global__ __launch_bounds__(256) void k_shared_up(const __bf16* __restrict__ xbf,
                                                   const float* __restrict__ ws1,
                                                   const float* __restrict__ ws3,
                                                   __bf16* __restrict__ hsbf) {
  __shared__ __bf16 As[2][64 * LDA];
  __shared__ __bf16 B1[2][128 * LDB];
  __shared__ __bf16 B3[2][128 * LDB];
  const int n0 = blockIdx.x * 128, m0 = blockIdx.y * 64;
  const int tid = threadIdx.x, lane = tid & 31, wave = tid >> 5;
  const int wy = wave >> 1, wx = wave & 1;

  const int ar = tid >> 2, ac8 = (tid & 3) * 8;
  const __bf16* aSrc = xbf + (size_t)(m0 + ar) * DIM + ac8;

  v8f zero = {};
  v8f acc1[4], acc3[4];
#pragma unroll
  for (int i = 0; i < 4; ++i) { acc1[i] = zero; acc3[i] = zero; }

  v4f q1[2][2], q3[2][2];
  auto loadB = [&](int k0) {
#pragma unroll
    for (int j = 0; j < 2; ++j) {
      const int id = tid + j * 256;
      const int kr = (id >> 5) * 2, c4 = (id & 31) * 4;
      const float* p1 = ws1 + (size_t)(k0 + kr) * HSH + n0 + c4;
      const float* p3 = ws3 + (size_t)(k0 + kr) * HSH + n0 + c4;
      q1[j][0] = *(const v4f*)p1;  q1[j][1] = *(const v4f*)(p1 + HSH);
      q3[j][0] = *(const v4f*)p3;  q3[j][1] = *(const v4f*)(p3 + HSH);
    }
  };
  auto storeB = [&](int buf) {
#pragma unroll
    for (int j = 0; j < 2; ++j) {
      const int id = tid + j * 256;
      const int kr = (id >> 5) * 2, c4 = (id & 31) * 4;
#pragma unroll
      for (int i = 0; i < 4; ++i) {
        *(unsigned*)(&B1[buf][(c4 + i) * LDB + kr]) = pack_bf16x2(q1[j][0][i], q1[j][1][i]);
        *(unsigned*)(&B3[buf][(c4 + i) * LDB + kr]) = pack_bf16x2(q3[j][0][i], q3[j][1][i]);
      }
    }
  };

  copy16_g2l(aSrc, &As[0][ar * LDA + ac8]);
  loadB(0);
  storeB(0);
  async_wait();
  __syncthreads();

  int cur = 0;
  for (int k0 = 0; k0 < DIM; k0 += 32) {
    const int nxt = cur ^ 1;
    const bool more = (k0 + 32) < DIM;
    if (more) {
      copy16_g2l(aSrc + k0 + 32, &As[nxt][ar * LDA + ac8]);
      loadB(k0 + 32);
    }
    v16bf a = load_a_frag(&As[cur][(wy * 16 + (lane & 15)) * LDA], lane);
#pragma unroll
    for (int nb = 0; nb < 4; ++nb) {
      const int col = wx * 64 + nb * 16 + (lane & 15);
      v16bf b1 = load_b_frag(&B1[cur][col * LDB], lane);
      v16bf b3 = load_b_frag(&B3[cur][col * LDB], lane);
      acc1[nb] = wmma_bf16(a, b1, acc1[nb]);
      acc3[nb] = wmma_bf16(a, b3, acc3[nb]);
    }
    if (more) storeB(nxt);
    async_wait();
    __syncthreads();
    cur = nxt;
  }

  const int rbase = m0 + wy * 16 + ((lane >> 4) << 3);
#pragma unroll
  for (int nb = 0; nb < 4; ++nb) {
    const int col = n0 + wx * 64 + nb * 16 + (lane & 15);
#pragma unroll
    for (int r = 0; r < 8; ++r) {
      const float g = silu_f(acc1[nb][r]) * acc3[nb][r];
      hsbf[(size_t)(rbase + r) * HSH + col] = (__bf16)g;
    }
  }
}

// ---------------- shared MLP down (double-buffered): out = 0.2*(hs@ws2) -----
__global__ __launch_bounds__(256) void k_shared_down(const __bf16* __restrict__ hsbf,
                                                     const float* __restrict__ ws2,
                                                     float* __restrict__ out) {
  __shared__ __bf16 As[2][64 * LDA];
  __shared__ __bf16 Bt[2][128 * LDB];
  const int n0 = blockIdx.x * 128, m0 = blockIdx.y * 64;
  const int tid = threadIdx.x, lane = tid & 31, wave = tid >> 5;
  const int wy = wave >> 1, wx = wave & 1;

  const int ar = tid >> 2, ac8 = (tid & 3) * 8;
  const __bf16* aSrc = hsbf + (size_t)(m0 + ar) * HSH + ac8;

  v8f zero = {};
  v8f acc[4];
#pragma unroll
  for (int i = 0; i < 4; ++i) acc[i] = zero;

  v4f q[2][2];
  auto loadB = [&](int k0) {
#pragma unroll
    for (int j = 0; j < 2; ++j) {
      const int id = tid + j * 256;
      const int kr = (id >> 5) * 2, c4 = (id & 31) * 4;
      const float* p = ws2 + (size_t)(k0 + kr) * DIM + n0 + c4;
      q[j][0] = *(const v4f*)p;  q[j][1] = *(const v4f*)(p + DIM);
    }
  };
  auto storeB = [&](int buf) {
#pragma unroll
    for (int j = 0; j < 2; ++j) {
      const int id = tid + j * 256;
      const int kr = (id >> 5) * 2, c4 = (id & 31) * 4;
#pragma unroll
      for (int i = 0; i < 4; ++i)
        *(unsigned*)(&Bt[buf][(c4 + i) * LDB + kr]) = pack_bf16x2(q[j][0][i], q[j][1][i]);
    }
  };

  copy16_g2l(aSrc, &As[0][ar * LDA + ac8]);
  loadB(0);
  storeB(0);
  async_wait();
  __syncthreads();

  int cur = 0;
  for (int k0 = 0; k0 < HSH; k0 += 32) {
    const int nxt = cur ^ 1;
    const bool more = (k0 + 32) < HSH;
    if (more) {
      copy16_g2l(aSrc + k0 + 32, &As[nxt][ar * LDA + ac8]);
      loadB(k0 + 32);
    }
    v16bf a = load_a_frag(&As[cur][(wy * 16 + (lane & 15)) * LDA], lane);
#pragma unroll
    for (int nb = 0; nb < 4; ++nb) {
      const int col = wx * 64 + nb * 16 + (lane & 15);
      v16bf b = load_b_frag(&Bt[cur][col * LDB], lane);
      acc[nb] = wmma_bf16(a, b, acc[nb]);
    }
    if (more) storeB(nxt);
    async_wait();
    __syncthreads();
    cur = nxt;
  }

  const int rbase = m0 + wy * 16 + ((lane >> 4) << 3);
#pragma unroll
  for (int nb = 0; nb < 4; ++nb) {
    const int col = n0 + wx * 64 + nb * 16 + (lane & 15);
#pragma unroll
    for (int r = 0; r < 8; ++r)
      out[(size_t)(rbase + r) * DIM + col] = 0.2f * acc[nb][r];
  }
}

// ---------------- fused sparse expert ---------------------------------------
__global__ __launch_bounds__(256) void k_expert(const __bf16* __restrict__ xbf,
                                                const float* __restrict__ we1,
                                                const float* __restrict__ we3,
                                                const float* __restrict__ we2,
                                                const float* __restrict__ cw,
                                                const int* __restrict__ counts,
                                                const int* __restrict__ tlist,
                                                float* __restrict__ out) {
  const int e    = blockIdx.y;
  const int tile = blockIdx.x;
  const int ntok = counts[e];
  if (tile * 64 >= ntok) return;   // uniform early exit (before any barrier)

  __shared__ __bf16 As[64 * LDA];
  __shared__ __bf16 Bt0[128 * LDB];
  __shared__ __bf16 Bt1[128 * LDB];
  __shared__ __bf16 G[64 * GLD];   // 64 x 1024 bf16 activation tile
  __shared__ int    toks[64];
  __shared__ float  cwr[64];

  const int tid = threadIdx.x, lane = tid & 31, wave = tid >> 5;
  const int wy = wave >> 1, wx = wave & 1;

  if (tid < 64) {
    const int idx = tile * 64 + tid;
    const int t = (idx < ntok) ? tlist[(size_t)e * TKN + idx]
                               : tlist[(size_t)e * TKN + tile * 64];
    toks[tid] = t;
    cwr[tid]  = (idx < ntok) ? cw[(size_t)t * NEXP + e] : 0.0f;  // pad rows -> g==0
  }
  __syncthreads();

  const float* W1 = we1 + (size_t)e * DIM * HEXP;
  const float* W3 = we3 + (size_t)e * DIM * HEXP;
  const float* W2 = we2 + (size_t)e * HEXP * DIM;

  const int ar = tid >> 2, ac8 = (tid & 3) * 8;
  const __bf16* aSrc = xbf + (size_t)toks[ar] * DIM + ac8;   // gathered row

  // ---- GEMM1: build g tile in LDS, 128 H-columns at a time ----
  v4f q1[2][2], q3[2][2];
  for (int h0 = 0; h0 < HEXP; h0 += 128) {
    v8f zero = {};
    v8f a1[4], a3[4];
#pragma unroll
    for (int i = 0; i < 4; ++i) { a1[i] = zero; a3[i] = zero; }

    for (int k0 = 0; k0 < DIM; k0 += 32) {
      copy16_g2l(aSrc + k0, &As[ar * LDA + ac8]);
#pragma unroll
      for (int j = 0; j < 2; ++j) {
        const int id = tid + j * 256;
        const int kr = (id >> 5) * 2, c4 = (id & 31) * 4;
        const float* p1 = W1 + (size_t)(k0 + kr) * HEXP + h0 + c4;
        const float* p3 = W3 + (size_t)(k0 + kr) * HEXP + h0 + c4;
        q1[j][0] = *(const v4f*)p1;  q1[j][1] = *(const v4f*)(p1 + HEXP);
        q3[j][0] = *(const v4f*)p3;  q3[j][1] = *(const v4f*)(p3 + HEXP);
      }
#pragma unroll
      for (int j = 0; j < 2; ++j) {
        const int id = tid + j * 256;
        const int kr = (id >> 5) * 2, c4 = (id & 31) * 4;
#pragma unroll
        for (int i = 0; i < 4; ++i) {
          *(unsigned*)(&Bt0[(c4 + i) * LDB + kr]) = pack_bf16x2(q1[j][0][i], q1[j][1][i]);
          *(unsigned*)(&Bt1[(c4 + i) * LDB + kr]) = pack_bf16x2(q3[j][0][i], q3[j][1][i]);
        }
      }
      async_wait();
      __syncthreads();
      v16bf a = load_a_frag(&As[(wy * 16 + (lane & 15)) * LDA], lane);
#pragma unroll
      for (int nb = 0; nb < 4; ++nb) {
        const int col = wx * 64 + nb * 16 + (lane & 15);
        v16bf b1 = load_b_frag(&Bt0[col * LDB], lane);
        v16bf b3 = load_b_frag(&Bt1[col * LDB], lane);
        a1[nb] = wmma_bf16(a, b1, a1[nb]);
        a3[nb] = wmma_bf16(a, b3, a3[nb]);
      }
      __syncthreads();
    }
    const int rloc = wy * 16 + ((lane >> 4) << 3);
#pragma unroll
    for (int nb = 0; nb < 4; ++nb) {
      const int col = h0 + wx * 64 + nb * 16 + (lane & 15);
#pragma unroll
      for (int r = 0; r < 8; ++r) {
        const int row = rloc + r;
        const float g = silu_f(a1[nb][r]) * a3[nb][r] * cwr[row];
        G[row * GLD + col] = (__bf16)g;
      }
    }
  }
  __syncthreads();

  // ---- GEMM2: out += 0.8 * (G @ we2[e]) ; B ping-pongs across Bt0/Bt1 ----
  v4f q2[2][2];
  auto loadB2 = [&](int n0, int k0) {
#pragma unroll
    for (int j = 0; j < 2; ++j) {
      const int id = tid + j * 256;
      const int kr = (id >> 5) * 2, c4 = (id & 31) * 4;
      const float* p = W2 + (size_t)(k0 + kr) * DIM + n0 + c4;
      q2[j][0] = *(const v4f*)p;  q2[j][1] = *(const v4f*)(p + DIM);
    }
  };
  auto storeB2 = [&](int buf) {
    __bf16* B = buf ? Bt1 : Bt0;     // runtime select (no LDS-pointer array)
#pragma unroll
    for (int j = 0; j < 2; ++j) {
      const int id = tid + j * 256;
      const int kr = (id >> 5) * 2, c4 = (id & 31) * 4;
#pragma unroll
      for (int i = 0; i < 4; ++i)
        *(unsigned*)(&B[(c4 + i) * LDB + kr]) = pack_bf16x2(q2[j][0][i], q2[j][1][i]);
    }
  };

  for (int n0 = 0; n0 < DIM; n0 += 128) {
    v8f zero = {};
    v8f acc[4];
#pragma unroll
    for (int i = 0; i < 4; ++i) acc[i] = zero;

    loadB2(n0, 0);
    storeB2(0);
    __syncthreads();

    int cur = 0;
    for (int k0 = 0; k0 < HEXP; k0 += 32) {
      const int nxt = cur ^ 1;
      const bool more = (k0 + 32) < HEXP;
      if (more) loadB2(n0, k0 + 32);
      const __bf16* Bc = cur ? Bt1 : Bt0;
      v16bf a = load_a_frag(&G[(wy * 16 + (lane & 15)) * GLD + k0], lane);
#pragma unroll
      for (int nb = 0; nb < 4; ++nb) {
        const int col = wx * 64 + nb * 16 + (lane & 15);
        v16bf b = load_b_frag(&Bc[col * LDB], lane);
        acc[nb] = wmma_bf16(a, b, acc[nb]);
      }
      if (more) storeB2(nxt);
      __syncthreads();
      cur = nxt;
    }

    const int rloc = wy * 16 + ((lane >> 4) << 3);
#pragma unroll
    for (int nb = 0; nb < 4; ++nb) {
      const int col = n0 + wx * 64 + nb * 16 + (lane & 15);
#pragma unroll
      for (int r = 0; r < 8; ++r) {
        const int row = rloc + r;
        atomicAdd(&out[(size_t)toks[row] * DIM + col], 0.8f * acc[nb][r]);
      }
    }
  }
}

// ---------------- launch ----------------
extern "C" void kernel_launch(void* const* d_in, const int* in_sizes, int n_in,
                              void* d_out, int out_size, void* d_ws, size_t ws_size,
                              hipStream_t stream) {
  const float* x   = (const float*)d_in[0];
  const float* wr  = (const float*)d_in[1];
  const float* we1 = (const float*)d_in[2];
  const float* we3 = (const float*)d_in[3];
  const float* we2 = (const float*)d_in[4];
  const float* ws1 = (const float*)d_in[5];
  const float* ws3 = (const float*)d_in[6];
  const float* ws2 = (const float*)d_in[7];
  float* out = (float*)d_out;

  char* w = (char*)d_ws;
  __bf16* xbf  = (__bf16*)w;  w += (size_t)TKN * DIM * 2;
  __bf16* hsbf = (__bf16*)w;  w += (size_t)TKN * HSH * 2;
  float*  cw   = (float*)w;   w += (size_t)TKN * NEXP * 4;
  int* counts  = (int*)w;     w += 256;
  int* tlist   = (int*)w;     w += (size_t)NEXP * TKN * 4;

  k_cvt_bf16<<<(TKN * DIM) / 2048, 256, 0, stream>>>(x, xbf);
  k_router<<<TKN, 32, 0, stream>>>(x, wr, cw);
  k_build_lists<<<NEXP, 256, 0, stream>>>(cw, counts, tlist);
  k_shared_up<<<dim3(HSH / 128, TKN / 64), 256, 0, stream>>>(xbf, ws1, ws3, hsbf);
  k_shared_down<<<dim3(DIM / 128, TKN / 64), 256, 0, stream>>>(hsbf, ws2, out);
  k_expert<<<dim3(TKN / 64, NEXP), 256, 0, stream>>>(xbf, we1, we3, we2,
                                                     cw, counts, tlist, out);
}
